// GraspConditionDeformableAttn_85985245266209
// MI455X (gfx1250) — matrix operand order repaired and checked by reading, not compiled, not run on hardware
//
#include <hip/hip_runtime.h>
#include <hip/hip_bf16.h>

#define FDIM 256
#define NH 4
#define EDIM 64
#define SPC 8       // SP^3
#define NSAMP 32    // SPC*NH samples per point
#define PTS 8       // points per workgroup
#define NS 4096

typedef __attribute__((ext_vector_type(16))) __bf16        v16bf;
typedef __attribute__((ext_vector_type(16))) unsigned short v16us;
typedef __attribute__((ext_vector_type(8)))  float          v8f;

// ---- LDS region map (byte offsets into one 152KB arena, time-disjoint) ----
#define SM_AUX    0        // [0,131072)      aux feats bf16; later K|V f32
#define SM_K      0        // [0,65536)       K f32 (phase >=4b)
#define SM_V      65536    // [65536,131072)  V f32
#define SM_FEAT   131072   // 8KB  f32        base feats (whole kernel)
#define SM_POS    139264   // 3KB  f32        aux positions (ph2-3)
#define SM_BUFK   139264   // 8KB  f32        Wk slice   (ph4 only)
#define SM_BUFV   147456   // 8KB  f32        Wv slice   (ph4 only)
#define SM_Q      139264   // 2KB  f32        queries    (ph5)
#define SM_SIM    141312   // 1KB  f32        sim/attn   (ph5)
#define SM_O      143360   // 8KB  bf16       attn out   (ph5c-6)
#define SM_WOBUF  0        // 32KB f32        Wo slice   (ph6, over dead K)
#define SM_BYTES  155648   // 152 KB total

__device__ __forceinline__ unsigned short f2bf(float f) {
    unsigned u = __builtin_bit_cast(unsigned, f);
    u = (u + 0x7FFFu + ((u >> 16) & 1u)) >> 16;   // round-to-nearest-even
    return (unsigned short)u;
}

// bilinear sample of one channel of a (256,64,64) plane at normalized (X,Y)
__device__ __forceinline__ float bilin(const float* __restrict__ plane, int c,
                                       float X, float Y) {
    float u = fminf(fmaxf(X, 0.0f), 1.0f) * 63.0f;
    float v = fminf(fmaxf(Y, 0.0f), 1.0f) * 63.0f;
    float u0 = floorf(u), v0 = floorf(v);
    int iu0 = (int)u0, iv0 = (int)v0;
    int iu1 = min(iu0 + 1, 63), iv1 = min(iv0 + 1, 63);
    float wu = u - u0, wv = v - v0;
    const float* pc = plane + c * 4096;
    float f00 = pc[iv0 * 64 + iu0], f01 = pc[iv0 * 64 + iu1];
    float f10 = pc[iv1 * 64 + iu0], f11 = pc[iv1 * 64 + iu1];
    float top = f00 + (f01 - f00) * wu;
    float bot = f10 + (f11 - f10) * wu;
    return top + (bot - top) * wv;
}

// Async DMA of 16 bytes/lane from global to LDS (ASYNCcnt-tracked).
__device__ __forceinline__ void async_b128(unsigned lds_dst_addr,
                                           int voff_bytes,
                                           const void* saddr) {
    asm volatile("global_load_async_to_lds_b128 %0, %1, %2"
                 :: "v"(lds_dst_addr), "v"(voff_bytes), "s"(saddr)
                 : "memory");
}
__device__ __forceinline__ void wait_async0() {
    asm volatile("s_wait_asynccnt 0x0" ::: "memory");
}

// A fragment: 16x32 bf16 tile from LDS (row-major, 256 halfs per row).
// A layout: lane L -> M=L%16; lanes 0-15 halves = K{0..15},
// lanes 16-31 halves = K{16..31} (matches B).
__device__ __forceinline__ v16bf loadA(const unsigned short* sA, int mtile,
                                       int kstep, int lane) {
    int M = mtile * 16 + (lane & 15);
    int kbase = kstep * 32 + ((lane & 16) ? 16 : 0);
    const unsigned short* r = sA + M * 256 + kbase;
    v16us u;
#pragma unroll
    for (int j = 0; j < 8; ++j) u[j] = r[j];
#pragma unroll
    for (int j = 0; j < 8; ++j) u[8 + j] = r[8 + j];
    return __builtin_bit_cast(v16bf, u);
}

// B fragment: 32x16 bf16 tile from an LDS-staged f32 slice (32 rows x ldn).
// B layout used here (contiguous k, same as A): lanes 0-15 -> N=lane hold K=0..15; lanes 16-31 -> K=16..31.
__device__ __forceinline__ v16bf loadBlds(const float* sB, int ntile, int ldn,
                                          int lane) {
    int n = ntile * 16 + (lane & 15);
    int kb = (lane & 16) ? 16 : 0;
    v16us u;
#pragma unroll
    for (int j = 0; j < 16; ++j) u[j] = f2bf(sB[(kb + j) * ldn + n]);
    return __builtin_bit_cast(v16bf, u);
}

__device__ __forceinline__ v8f wmma_bf16(v16bf a, v16bf b, v8f c) {
    return __builtin_amdgcn_wmma_f32_16x16x32_bf16(false, a, false, b,
                                                   (short)0, c, false, false);
}

__launch_bounds__(256)
__global__ void grasp_attn_kernel(
    const float* __restrict__ grasp, const float* __restrict__ pxy,
    const float* __restrict__ pxz,   const float* __restrict__ pyz,
    const float* __restrict__ Wq, const float* __restrict__ bq,
    const float* __restrict__ Wk, const float* __restrict__ bk,
    const float* __restrict__ Wv, const float* __restrict__ bv,
    const float* __restrict__ Wo, const float* __restrict__ bo,
    const float* __restrict__ Woff, const float* __restrict__ boff,
    float* __restrict__ out)
{
    __shared__ __align__(16) unsigned char smem[SM_BYTES];

    unsigned short* sAux = (unsigned short*)(smem + SM_AUX);
    float* sK    = (float*)(smem + SM_K);
    float* sV    = (float*)(smem + SM_V);
    float* sFeat = (float*)(smem + SM_FEAT);
    float* sPos  = (float*)(smem + SM_POS);
    float* sBufK = (float*)(smem + SM_BUFK);
    float* sBufV = (float*)(smem + SM_BUFV);
    float* sQ    = (float*)(smem + SM_Q);
    float* sSim  = (float*)(smem + SM_SIM);
    unsigned short* sO = (unsigned short*)(smem + SM_O);
    float* sWoBuf = (float*)(smem + SM_WOBUF);

    const int tid  = threadIdx.x;
    const int lane = tid & 31;
    const int wave = tid >> 5;
    const int pbase = blockIdx.x * PTS;
    const int batch = pbase / NS;                  // 8 pts share a batch
    const float* PXY = pxy + (size_t)batch * FDIM * 4096;
    const float* PXZ = pxz + (size_t)batch * FDIM * 4096;
    const float* PYZ = pyz + (size_t)batch * FDIM * 4096;

    // touch weight matrices into cache (global_prefetch)
    __builtin_prefetch(Wk + tid * 64, 0, 1);
    __builtin_prefetch(Wv + tid * 64, 0, 1);
    __builtin_prefetch(Wo + tid * 256, 0, 1);

    // ---- Phase 1: base features: thread = channel, loop points ----
    for (int p = 0; p < PTS; ++p) {
        const float* g = grasp + (size_t)(pbase + p) * 7;
        float x = g[0], y = g[1], z = g[2];
        float f = bilin(PXY, tid, x, y) + bilin(PXZ, tid, x, z) +
                  bilin(PYZ, tid, y, z);
        sFeat[p * FDIM + tid] = f;
    }
    __syncthreads();

    // ---- Phase 2: offset MLP (fc @ Woff + boff) -> aux positions ----
#pragma unroll
    for (int it = 0; it < 3; ++it) {               // 768 tasks
        int task = it * 256 + tid;
        int p = task / 96;
        int o = task % 96;
        const float* g = grasp + (size_t)(pbase + p) * 7;
        float acc = boff[o];
        const float* wcol = Woff + o;
        const float* fr = sFeat + p * FDIM;
        for (int i = 0; i < FDIM; ++i) acc += fr[i] * wcol[(size_t)i * 96];
#pragma unroll
        for (int i = 0; i < 4; ++i) acc += g[3 + i] * wcol[(size_t)(FDIM + i) * 96];
        int s = o / 3, d = o % 3;                  // s = j*NH + h
        sPos[(p * NSAMP + s) * 3 + d] = acc + g[d];
    }
    __syncthreads();

    // ---- Phase 3: aux features (A matrix, 256 rows x 256 K, bf16) ----
    for (int r = 0; r < PTS * NSAMP; ++r) {
        float x = sPos[r * 3 + 0], y = sPos[r * 3 + 1], z = sPos[r * 3 + 2];
        float f = bilin(PXY, tid, x, y) + bilin(PXZ, tid, x, z) +
                  bilin(PYZ, tid, y, z);
        sAux[r * FDIM + tid] = f2bf(f);
    }
    __syncthreads();

    // ---- Phase 4: K and V projections via WMMA, async-staged B slices ----
    // K-step outer loop; all 8 waves share one 32x64 slice of Wk and Wv,
    // staged per step with global_load_async_to_lds_b128 (ASYNCcnt).
    v8f accK[8], accV[8];
#pragma unroll
    for (int t = 0; t < 8; ++t) {
        accK[t] = (v8f){0, 0, 0, 0, 0, 0, 0, 0};
        accV[t] = (v8f){0, 0, 0, 0, 0, 0, 0, 0};
    }
    for (int ks = 0; ks < 8; ++ks) {
        {   // threads 0..127 stage Wk slice, 128..255 stage Wv slice (8KB ea)
            const float* src = (tid < 128) ? Wk : Wv;
            unsigned dbase = (unsigned)(uintptr_t)((tid < 128) ? sBufK : sBufV);
            const float* sbase = src + ks * 32 * EDIM;
            int lt = tid & 127;
#pragma unroll
            for (int j = 0; j < 4; ++j) {
                int off = lt * 16 + j * 2048;
                async_b128(dbase + (unsigned)off, off, sbase);
            }
        }
        wait_async0();
        __syncthreads();
#pragma unroll
        for (int t = 0; t < 8; ++t) {
            int T = wave + 8 * t;
            int m = T >> 2, n = T & 3;
            v16bf a = loadA(sAux, m, ks, lane);          // reused for K and V
            accK[t] = wmma_bf16(a, loadBlds(sBufK, n, EDIM, lane), accK[t]);
            accV[t] = wmma_bf16(a, loadBlds(sBufV, n, EDIM, lane), accV[t]);
        }
        __syncthreads();                                 // before re-staging
    }
    // all sAux reads complete (loop-final barrier): overwrite with K|V f32
#pragma unroll
    for (int t = 0; t < 8; ++t) {
        int T = wave + 8 * t;
        int m = T >> 2, n = T & 3;
        int col   = n * 16 + (lane & 15);
        int rbase = m * 16 + ((lane & 16) ? 8 : 0);
#pragma unroll
        for (int r = 0; r < 8; ++r) {
            sK[(rbase + r) * EDIM + col] = accK[t][r] + bk[col];
            sV[(rbase + r) * EDIM + col] = accV[t][r] + bv[col];
        }
    }
    __syncthreads();

    // ---- Phase 5a: q = (feat @ Wq + bq) * sqrt(EDIM) ----
#pragma unroll
    for (int it = 0; it < 2; ++it) {               // 512 tasks
        int task = it * 256 + tid;
        int p = task >> 6, d = task & 63;
        float acc = bq[d];
        const float* fr = sFeat + p * FDIM;
        for (int i = 0; i < FDIM; ++i) acc += fr[i] * Wq[i * EDIM + d];
        sQ[p * EDIM + d] = acc * 8.0f;             // EDIM**0.5
    }
    __syncthreads();

    // ---- Phase 5b: sim[p][h][j] = q . k ----
    {
        int p = tid >> 5, h = (tid >> 3) & 3, j = tid & 7;
        const float* qv = sQ + p * EDIM;
        const float* kv = sK + (p * NSAMP + j * NH + h) * EDIM;
        float s = 0.0f;
        for (int d = 0; d < EDIM; ++d) s += qv[d] * kv[d];
        sSim[tid] = s;                             // tid == (p*4+h)*8+j
    }
    __syncthreads();
    if (tid < 32) {                                // softmax over j (8 vals)
        float* row = sSim + tid * 8;
        float m = row[0];
#pragma unroll
        for (int j = 1; j < 8; ++j) m = fmaxf(m, row[j]);
        float e[8], sum = 0.0f;
#pragma unroll
        for (int j = 0; j < 8; ++j) { e[j] = expf(row[j] - m); sum += e[j]; }
        float inv = 1.0f / sum;
#pragma unroll
        for (int j = 0; j < 8; ++j) row[j] = e[j] * inv;
    }
    __syncthreads();

    // ---- Phase 5c: o = attn @ V -> bf16 A-matrix (rows 8..15 zero) ----
#pragma unroll
    for (int it = 0; it < 8; ++it) {               // 2048 tasks
        int task = it * 256 + tid;
        int p = task >> 8, ch = task & 255;
        int h = ch >> 6, d = ch & 63;
        const float* at = sSim + (p * NH + h) * SPC;
        float acc = 0.0f;
#pragma unroll
        for (int j = 0; j < 8; ++j)
            acc += at[j] * sV[(p * NSAMP + j * NH + h) * EDIM + d];
        sO[p * FDIM + ch] = f2bf(acc);
        sO[(p + 8) * FDIM + ch] = 0;               // zero pad
    }
    __syncthreads();

    // ---- Phase 6: out = o @ Wo + bo + feat, async-staged Wo slices ----
    // 32KB slice of Wo per K-step staged into LDS (over dead K region).
    v8f acc0 = {0, 0, 0, 0, 0, 0, 0, 0};
    v8f acc1 = {0, 0, 0, 0, 0, 0, 0, 0};
    for (int ks = 0; ks < 8; ++ks) {
        const float* sbase = Wo + ks * 32 * FDIM;  // 8192 floats = 32KB
        unsigned dbase = (unsigned)(uintptr_t)sWoBuf;
#pragma unroll
        for (int j = 0; j < 8; ++j) {
            int off = tid * 16 + j * 4096;
            async_b128(dbase + (unsigned)off, off, sbase);
        }
        wait_async0();
        __syncthreads();
        v16bf a = loadA(sO, 0, ks, lane);
        acc0 = wmma_bf16(a, loadBlds(sWoBuf, wave * 2 + 0, FDIM, lane), acc0);
        acc1 = wmma_bf16(a, loadBlds(sWoBuf, wave * 2 + 1, FDIM, lane), acc1);
        __syncthreads();
    }
#pragma unroll
    for (int t = 0; t < 2; ++t) {
        v8f acc = (t == 0) ? acc0 : acc1;
        if (lane < 16) {                           // lanes 0-15 hold rows 0..7
            int col = (wave * 2 + t) * 16 + lane;
            float bias = bo[col];
#pragma unroll
            for (int r = 0; r < 8; ++r)
                out[(size_t)(pbase + r) * FDIM + col] =
                    acc[r] + bias + sFeat[r * FDIM + col];
        }
    }
}

extern "C" void kernel_launch(void* const* d_in, const int* in_sizes, int n_in,
                              void* d_out, int out_size, void* d_ws, size_t ws_size,
                              hipStream_t stream) {
    (void)in_sizes; (void)n_in; (void)d_ws; (void)ws_size; (void)out_size;
    const float* grasp = (const float*)d_in[0];
    const float* pxy   = (const float*)d_in[1];
    const float* pxz   = (const float*)d_in[2];
    const float* pyz   = (const float*)d_in[3];
    const float* Wq    = (const float*)d_in[4];
    const float* bq    = (const float*)d_in[5];
    const float* Wk    = (const float*)d_in[6];
    const float* bk    = (const float*)d_in[7];
    const float* Wv    = (const float*)d_in[8];
    const float* bv    = (const float*)d_in[9];
    const float* Wo    = (const float*)d_in[10];
    const float* bo    = (const float*)d_in[11];
    const float* Woff  = (const float*)d_in[12];
    const float* boff  = (const float*)d_in[13];
    float* out = (float*)d_out;

    dim3 grid((4 * NS) / PTS);                     // 2048 workgroups
    hipLaunchKernelGGL(grasp_attn_kernel, grid, dim3(256), 0, stream,
                       grasp, pxy, pxz, pyz, Wq, bq, Wk, bk, Wv, bv,
                       Wo, bo, Woff, boff, out);
}
